// RNN_40106404610602
// MI455X (gfx1250) — matrix operand order repaired and checked by reading, not compiled, not run on hardware
//
#include <hip/hip_runtime.h>
#include <hip/hip_bf16.h>

typedef __attribute__((ext_vector_type(16))) __bf16 v16bf;
typedef __attribute__((ext_vector_type(8)))  float  v8f;
typedef __attribute__((ext_vector_type(4)))  unsigned int v4u;
typedef __attribute__((ext_vector_type(8)))  int    v8i;
typedef __attribute__((ext_vector_type(4)))  int    v4i;

#define VOCAB 32000
#define HID   512
#define BATCH 16
#define SEQ   128
#define NT_Q  (VOCAB / 16)   // 2000 projection N-tiles
#define NT_H  (HID / 16)     // 32 recurrence N-tiles
#define KSTEPS (HID / 32)    // 16 k-steps of depth 32

// ---- CDNA5 feature guards (device pass only; host parse takes fallbacks) ----
#if defined(__HIP_DEVICE_COMPILE__) && __has_builtin(__builtin_amdgcn_tensor_load_to_lds) && __has_builtin(__builtin_amdgcn_s_wait_tensorcnt)
#define USE_TDM 1
#else
#define USE_TDM 0
#endif

#if defined(__HIP_DEVICE_COMPILE__) && __has_builtin(__builtin_amdgcn_tanhf)
#define TANH_F32 __builtin_amdgcn_tanhf
#elif defined(__HIP_DEVICE_COMPILE__) && __has_builtin(__builtin_amdgcn_tanh_f32)
#define TANH_F32 __builtin_amdgcn_tanh_f32
#else
#define TANH_F32 tanhf
#endif

// D = A(16x32 bf16) * B(32x16 bf16) + C(16x16 f32), wave32
__device__ __forceinline__ v8f wmma_bf16(v16bf a, v16bf b, v8f c) {
    return __builtin_amdgcn_wmma_f32_16x16x32_bf16(
        false, a, false, b, (short)0, c, false, false);
}

#if USE_TDM
// One 2D TDM descriptor: 8 rows x 1024B, global row stride 16KB -> contiguous LDS.
// data_size=8B: tile_dim0 = tensor_dim0 = 128, tile_dim1 = tensor_dim1 = 8,
// tensor_dim0_stride = 2048 (16KB/8B). Issued by one wave; tracked on TENSORcnt.
__device__ __forceinline__ void tdm_load_slab(const void* gptr, void* lptr) {
    unsigned long ga = (unsigned long)gptr;
    unsigned      la = (unsigned)(unsigned long)lptr;   // addr[31:0] == LDS byte offset
    v4u g0 = { 1u,                                      // count=1, is_restore=0
               la,                                      // lds_addr
               (unsigned)ga,                            // global_addr[31:0]
               (unsigned)((ga >> 32) & 0x1FFFFFFu) | (2u << 30) }; // ga[56:32] | type=2
    v8i g1 = { (3 << 16),      // data_size = 3 (8 bytes)
               (128 << 16),    // tensor_dim0 = 128 (bits 79:48)
               (8  << 16),     // tensor_dim1 = 8   (bits 111:80)
               (128 << 16),    // tile_dim0   = 128 (bits 127:112)
               8,              // tile_dim1   = 8   (bits 143:128)
               2048,           // tensor_dim0_stride[31:0]
               0, 0 };
    v4i gz = { 0, 0, 0, 0 };
#if __clang_major__ >= 23
    v8i gz8 = { 0, 0, 0, 0, 0, 0, 0, 0 };
    __builtin_amdgcn_tensor_load_to_lds(g0, g1, gz, gz, gz8, 0);
#else
    __builtin_amdgcn_tensor_load_to_lds(g0, g1, gz, gz, 0);
#endif
}
#endif

// ---------------------------------------------------------------------------
// Pack f32 weights (row-major K x N) into bf16 WMMA B-operand tiles.
// Slot s = (n_tile*KSTEPS + k_block)*32 + lane; element i of lane L maps to
//   K = k_block*32 + (i/8)*16 + (L/16)*8 + (i%8),  N = n_tile*16 + (L%16)
// ---------------------------------------------------------------------------
__global__ void pack_b_kernel(const float* __restrict__ Whq,
                              const float* __restrict__ Whh,
                              __bf16* __restrict__ whq_p,
                              __bf16* __restrict__ whh_p) {
    long tid = (long)blockIdx.x * blockDim.x + threadIdx.x;
    const long nq = (long)NT_Q * KSTEPS * 32;
    const float* src;
    __bf16* dst;
    int ld;
    long s;
    if (tid < nq) { src = Whq; dst = whq_p; ld = VOCAB; s = tid; }
    else {
        s = tid - nq;
        if (s >= (long)NT_H * KSTEPS * 32) return;
        src = Whh; dst = whh_p; ld = HID;
    }
    int L  = (int)(s & 31);
    int kb = (int)((s >> 5) & (KSTEPS - 1));
    int nt = (int)(s / (32 * KSTEPS));
    int kbase = kb * 32 + (L >> 4) * 8;
    int N = nt * 16 + (L & 15);
    v16bf v;
#pragma unroll
    for (int i = 0; i < 16; i++)
        v[i] = (__bf16)src[(long)(kbase + (i >> 3) * 16 + (i & 7)) * ld + N];
    *(v16bf*)(dst + s * 16) = v;
}

// ---------------------------------------------------------------------------
// Phase 1: sequential recurrence, one workgroup of 512 threads (16 waves).
// h_t (16x512) lives in LDS in WMMA A-operand layout (16 KB). Per step:
// 32x16 bf16 WMMAs on h_{t-1}, barrier, fused gather(Wxh[x])+bias+v_tanh,
// transpose-scatter h_t to LDS, barrier, coalesced 16KB copy to a_packed[t].
// ---------------------------------------------------------------------------
__global__ __launch_bounds__(512, 1) void rnn_recur_kernel(
    const int* __restrict__ x, const float* __restrict__ state,
    const float* __restrict__ Wxh, const float* __restrict__ bh,
    const __bf16* __restrict__ whh_p, __bf16* __restrict__ a_packed,
    float* __restrict__ h_final_out) {
    __shared__ v16bf lds_h[512];            // 16 KB: h in A-operand layout
    __shared__ int   lds_x[BATCH * SEQ];    // 8 KB token ids

    const int tid = threadIdx.x;
    const int L   = tid & 31;
    const int w   = tid >> 5;

    for (int j = tid; j < BATCH * SEQ; j += 512) lds_x[j] = x[j];

    {   // initial h = state, converted into A-layout
        int kb = tid >> 5, Ld = tid & 31;
        int M = Ld & 15, hi = Ld >> 4;
        v16bf v;
#pragma unroll
        for (int i = 0; i < 16; i++) {
            int Kp = (i >> 3) * 16 + hi * 8 + (i & 7);
            v[i] = (__bf16)state[M * HID + kb * 32 + Kp];
        }
        lds_h[tid] = v;
    }
    __syncthreads();

    const int nt0 = 2 * w, nt1 = 2 * w + 1;
    const int gn0 = nt0 * 16 + (L & 15);
    const int gn1 = nt1 * 16 + (L & 15);
    const float bh0 = bh[gn0], bh1 = bh[gn1];
    __bf16* lds_sc = (__bf16*)lds_h;

    for (int t = 0; t < SEQ; t++) {
        v8f acc0 = {}, acc1 = {};
#pragma unroll 4
        for (int ks = 0; ks < KSTEPS; ks++) {
            v16bf a  = lds_h[ks * 32 + L];
            v16bf b0 = *(const v16bf*)(whh_p + ((long)(nt0 * KSTEPS + ks) * 32 + L) * 16);
            v16bf b1 = *(const v16bf*)(whh_p + ((long)(nt1 * KSTEPS + ks) * 32 + L) * 16);
            acc0 = wmma_bf16(a, b0, acc0);
            acc1 = wmma_bf16(a, b1, acc1);
        }
        __syncthreads();    // everyone done reading h_{t-1}
#pragma unroll
        for (int r = 0; r < 8; r++) {
            int m   = r + (L >> 4) * 8;
            int tok = lds_x[m * SEQ + t];
            float h0 = TANH_F32(acc0[r] + Wxh[(long)tok * HID + gn0] + bh0);
            float h1 = TANH_F32(acc1[r] + Wxh[(long)tok * HID + gn1] + bh1);
            {
                int Kp = gn0 & 31, kb = gn0 >> 5;
                int Ld = m + 16 * ((Kp & 15) >> 3);
                int ii = (Kp >> 4) * 8 + (Kp & 7);
                lds_sc[(kb * 32 + Ld) * 16 + ii] = (__bf16)h0;
            }
            {
                int Kp = gn1 & 31, kb = gn1 >> 5;
                int Ld = m + 16 * ((Kp & 15) >> 3);
                int ii = (Kp >> 4) * 8 + (Kp & 7);
                lds_sc[(kb * 32 + Ld) * 16 + ii] = (__bf16)h1;
            }
            if (t == SEQ - 1) {
                h_final_out[m * HID + gn0] = h0;
                h_final_out[m * HID + gn1] = h1;
            }
        }
        __syncthreads();    // h_t fully in LDS
        *(v16bf*)(a_packed + ((long)(t * 512 + tid) * 16)) = lds_h[tid];
    }
}

// ---------------------------------------------------------------------------
// Phase 2: Y = H_all(2048x512) @ Whq(512x32000) + bq.
// 128x128 output block per workgroup (8 waves x [4 M-tiles x 2 N-tiles]).
// Per k-step, A slab (8x1KB) and B slab (8x1KB) are TDM'd into LDS double
// buffers (wave 0 issues, prefetched one k-step ahead, s_wait_tensorcnt),
// compute waves read only LDS. 128 WMMAs per wave. Nontemporal f32 stores
// keep the 262 MB output stream from thrashing the L2-resident weights.
// ---------------------------------------------------------------------------
__global__ __launch_bounds__(256) void rnn_proj_kernel(
    const __bf16* __restrict__ whq_p, const __bf16* __restrict__ a_packed,
    const float* __restrict__ bq, float* __restrict__ out) {
    __shared__ v16bf bufA[2][256];   // 2 x 8KB
    __shared__ v16bf bufB[2][256];   // 2 x 8KB

    const int tid = threadIdx.x;
    const int L   = tid & 31;
    const int w   = tid >> 5;
    const int nb  = blockIdx.x / 16;     // N-block: 128 cols   (0..249)
    const int mb  = blockIdx.x % 16;     // M-block: 128 rows   (0..15)
    const int mh  = w & 1;               // M half   (4 M-tiles each)
    const int np  = w >> 1;              // N pair   (2 N-tiles each)

    const char* aBase = (const char*)a_packed + (long)(mb * 8) * 16 * 1024;
    const char* bBase = (const char*)whq_p   + (long)(nb * 8) * 16 * 1024;

    v8f acc[4][2] = {{{}, {}}, {{}, {}}, {{}, {}}, {{}, {}}};

#if USE_TDM
    if (w == 0) {   // prologue: stage k-step 0
        tdm_load_slab(aBase, &bufA[0][0]);
        tdm_load_slab(bBase, &bufB[0][0]);
    }
#endif
    for (int ks = 0; ks < KSTEPS; ks++) {
#if USE_TDM
        const int buf = ks & 1;
        if (w == 0) {
            if (ks < KSTEPS - 1) {      // prefetch next slab pair
                tdm_load_slab(aBase + (ks + 1) * 1024, &bufA[buf ^ 1][0]);
                tdm_load_slab(bBase + (ks + 1) * 1024, &bufB[buf ^ 1][0]);
                __builtin_amdgcn_s_wait_tensorcnt(2);   // current pair done
            } else {
                __builtin_amdgcn_s_wait_tensorcnt(0);
            }
        }
        __syncthreads();                // slab ks visible to all waves
#else
        const int buf = 0;
        __syncthreads();                // prev reads done before overwrite
        {   // cooperative copy fallback: 256 threads x 32B per slab
            int row = tid >> 5;
            bufA[0][row * 32 + L] = *(const v16bf*)(aBase + (long)row * 16 * 1024 + (long)ks * 1024 + L * 32);
            bufB[0][row * 32 + L] = *(const v16bf*)(bBase + (long)row * 16 * 1024 + (long)ks * 1024 + L * 32);
        }
        __syncthreads();
#endif
        v16bf b0 = bufB[buf][(np * 2 + 0) * 32 + L];
        v16bf b1 = bufB[buf][(np * 2 + 1) * 32 + L];
#pragma unroll
        for (int q = 0; q < 4; q++) {
            v16bf a = bufA[buf][(mh * 4 + q) * 32 + L];
            acc[q][0] = wmma_bf16(a, b0, acc[q][0]);
            acc[q][1] = wmma_bf16(a, b1, acc[q][1]);
        }
#if USE_TDM
        __syncthreads();                // reads of buf done before its reuse
#endif
    }

    int gn0 = nb * 128 + np * 32 + (L & 15);
    int gn1 = gn0 + 16;
    float bq0 = bq[gn0], bq1 = bq[gn1];
#pragma unroll
    for (int q = 0; q < 4; q++) {
#pragma unroll
        for (int r = 0; r < 8; r++) {
            long M = (long)(mb * 8 + mh * 4 + q) * 16 + r + (L >> 4) * 8;
            __builtin_nontemporal_store(acc[q][0][r] + bq0, out + M * VOCAB + gn0);
            __builtin_nontemporal_store(acc[q][1][r] + bq1, out + M * VOCAB + gn1);
        }
    }
}

extern "C" void kernel_launch(void* const* d_in, const int* in_sizes, int n_in,
                              void* d_out, int out_size, void* d_ws, size_t ws_size,
                              hipStream_t stream) {
    const int*   x     = (const int*)d_in[0];
    const float* state = (const float*)d_in[1];
    const float* Wxh   = (const float*)d_in[2];
    const float* Whh   = (const float*)d_in[3];
    const float* bh    = (const float*)d_in[4];
    const float* Whq   = (const float*)d_in[5];
    const float* bq    = (const float*)d_in[6];
    float* out = (float*)d_out;

    char* ws = (char*)d_ws;
    __bf16* whq_p = (__bf16*)(ws);                // 32,768,000 B
    __bf16* whh_p = (__bf16*)(ws + 32768000);     //    524,288 B
    __bf16* a_p   = (__bf16*)(ws + 33292288);     //  2,097,152 B

    pack_b_kernel<<<4064, 256, 0, stream>>>(Whq, Whh, whq_p, whh_p);
    rnn_recur_kernel<<<1, 512, 0, stream>>>(x, state, Wxh, bh, whh_p, a_p,
                                            out + 65536000L);
    rnn_proj_kernel<<<4000, 256, 0, stream>>>(whq_p, a_p, bq, out);
}
